// FNO3d_onestep_72387378807359
// MI455X (gfx1250) — matrix operand order repaired and checked by reading, not compiled, not run on hardware
//
#include <hip/hip_runtime.h>
#include <hip/hip_bf16.h>
#include <stdint.h>

// ---------------------------------------------------------------------------
// FNO3d one-step for MI455X (gfx1250, wave32, WMMA + TDM).
// Activations: h[b][c][x][y][z], z fastest. P = 64^3.
// Spectral path: truncated separable DFTs (16x16x8 corner modes) with a shared
// 64-entry LDS twiddle table; mode-mix and pointwise conv on
// v_wmma_f32_16x16x32_f16; inverse-z DFT fused into the pointwise epilogue;
// forward z-DFT stages its input rows into LDS via the Tensor Data Mover.
// ---------------------------------------------------------------------------

typedef _Float16 v16h __attribute__((ext_vector_type(16)));
typedef float    v8f  __attribute__((ext_vector_type(8)));

#define TWOPI_64 0.09817477042468103f   // 2*pi/64
#define PVOL 262144L                     // 64^3

#ifndef __has_builtin
#define __has_builtin(x) 0
#endif

#if __has_builtin(__builtin_amdgcn_tensor_load_to_lds) && \
    __has_builtin(__builtin_amdgcn_s_wait_tensorcnt)
#define FNO_HAS_TDM 1
#else
#define FNO_HAS_TDM 0
#endif

#if FNO_HAS_TDM
typedef unsigned int u32x4 __attribute__((ext_vector_type(4)));
typedef int          i32x4 __attribute__((ext_vector_type(4)));
typedef int          i32x8 __attribute__((ext_vector_type(8)));

// TDM: DMA n_f32 contiguous floats from global memory into LDS.
// D# per CDNA5 ISA ch.8: group0 = {count, lds_addr, global_addr, type=2},
// group1 = {mask=0, data_size=4B, dims/strides}, groups 2/3 unused (2-D).
__device__ __forceinline__ void tdm_load_f32(const float* gsrc, void* lds_dst,
                                             int n_f32) {
    unsigned lds_addr = (unsigned)(uintptr_t)lds_dst;   // LDS aperture low 32b
    unsigned long long ga = (unsigned long long)(uintptr_t)gsrc;
    u32x4 g0;
    g0.x = 1u;                                          // count=1, user D#
    g0.y = lds_addr;                                    // lds_addr [63:32]
    g0.z = (unsigned)(ga & 0xFFFFFFFFull);              // global_addr [95:64]
    g0.w = (unsigned)((ga >> 32) & 0x01FFFFFFull)       // global_addr [120:96]
         | (2u << 30);                                  // type=2 ("image")
    i32x8 g1;
    g1[0] = (2 << 16);                                  // data_size=4B, mask=0
    g1[1] = (n_f32 & 0xFFFF) << 16;                     // tensor_dim0 lo16
    g1[2] = ((n_f32 >> 16) & 0xFFFF) | (1 << 16);       // dim0 hi16, dim1=1
    g1[3] = (n_f32 & 0xFFFF) << 16;                     // tile_dim0
    g1[4] = 1;                                          // tile_dim1=1
    g1[5] = n_f32;                                      // dim0_stride lo32
    g1[6] = 0;
    g1[7] = 0;
    i32x4 gz = {0, 0, 0, 0};
#if defined(__clang_major__) && (__clang_major__ >= 23)
    i32x8 gz8 = {0, 0, 0, 0, 0, 0, 0, 0};
    __builtin_amdgcn_tensor_load_to_lds(g0, g1, gz, gz, gz8, 0);
#else
    __builtin_amdgcn_tensor_load_to_lds(g0, g1, gz, gz, 0);
#endif
}
#endif  // FNO_HAS_TDM

// A-matrix (16x32 f16) per-lane K index for element e (ISA 7.12.2).
__device__ __forceinline__ int kmapA(int e, int hi) {
    int p = e >> 1, c = e & 1;
    return (p < 4) ? (hi * 8 + p * 2 + c) : (16 + hi * 8 + (p - 4) * 2 + c);
}
// B-matrix (32x16 f16): same K order as A.
__device__ __forceinline__ int kmapB(int e, int hi) { return (e < 8) ? (hi * 8 + e) : (16 + hi * 8 + (e - 8)); }

__device__ __forceinline__ v8f wmma16(const v16h& a, const v16h& b, const v8f& c) {
    return __builtin_amdgcn_wmma_f32_16x16x32_f16(false, a, false, b, (short)0, c,
                                                  false, false);
}

// Shared twiddle-table initializer: tc[t]=cos(2*pi*t/64), ts[t]=sin(2*pi*t/64).
__device__ __forceinline__ void init_twiddle(float* tc, float* ts) {
    int t = threadIdx.x;
    if (t < 64) {
        tc[t] = cosf(TWOPI_64 * (float)t);
        ts[t] = sinf(TWOPI_64 * (float)t);
    }
}

// ---------------------------------------------------------------------------
// 1) Lift: h[b][co][p] = p_b[co] + sum_j f_j * p_w[j][co], f = [x, gx, gy, gz]
// ---------------------------------------------------------------------------
__global__ __launch_bounds__(256) void fno_lift(const float* __restrict__ xin,
                                                const float* __restrict__ p_w,
                                                const float* __restrict__ p_b,
                                                float* __restrict__ h) {
    long t = (long)blockIdx.x * 256 + threadIdx.x;   // 0 .. B*P-1
    long b = t >> 18;
    long p = t & (PVOL - 1);
    int xi = (int)(p >> 12), yi = (int)((p >> 6) & 63), zi = (int)(p & 63);
    float f0 = xin[t];
    float f1 = xi * (1.0f / 63.0f);
    float f2 = yi * (1.0f / 63.0f);
    float f3 = zi * (1.0f / 63.0f);
    float* hb = h + b * 32 * PVOL + p;
#pragma unroll
    for (int co = 0; co < 32; ++co) {
        float acc = p_b[co]
                  + f0 * p_w[co] + f1 * p_w[32 + co]
                  + f2 * p_w[64 + co] + f3 * p_w[96 + co];
        hb[(long)co * PVOL] = acc;
    }
}

// ---------------------------------------------------------------------------
// 2) Forward truncated DFT along z: Z[bc][x][y][kz<8] = sum_z h*e^{-i th kz z}
//    Each block handles 32 rows x 8 kz. The 8 KB row tile is staged into LDS
//    by one TDM descriptor (TENSORcnt), so each row is fetched from memory
//    once instead of 8x; twiddles via incremental (id + kz) & 63.
// ---------------------------------------------------------------------------
__global__ __launch_bounds__(256) void fno_dftz_fwd(const float* __restrict__ h,
                                                    float* __restrict__ Zre,
                                                    float* __restrict__ Zim) {
    __shared__ float tc[64], ts[64];
    __shared__ float tile[2048];                     // 32 rows x 64 z
    init_twiddle(tc, ts);

    const float* gsrc = h + (long)blockIdx.x * 2048;
#if FNO_HAS_TDM
    if (threadIdx.x < 32) {                          // one wave issues the DMA
        tdm_load_f32(gsrc, tile, 2048);
        __builtin_amdgcn_s_wait_tensorcnt(0);
    }
#else
    for (int i = threadIdx.x; i < 2048; i += 256) tile[i] = gsrc[i];
#endif
    __syncthreads();

    long t = (long)blockIdx.x * 256 + threadIdx.x;   // NBC*4096*8
    int kz = (int)(t & 7);
    const float* hp = tile + (threadIdx.x >> 3) * 64;
    float sr = 0.f, si = 0.f;
    int id = 0;
#pragma unroll 16
    for (int z = 0; z < 64; ++z) {
        float v = hp[z];
        sr += v * tc[id];
        si -= v * ts[id];         // e^{-i th} = cos - i sin
        id = (id + kz) & 63;
    }
    Zre[t] = sr; Zim[t] = si;
}

// ---------------------------------------------------------------------------
// 3) Forward DFT along y (16 kept modes): Y[bc][x][ky][kz]
// ---------------------------------------------------------------------------
__global__ __launch_bounds__(256) void fno_dfty_fwd(const float* __restrict__ Zre,
                                                    const float* __restrict__ Zim,
                                                    float* __restrict__ Yre,
                                                    float* __restrict__ Yim) {
    __shared__ float tc[64], ts[64];
    init_twiddle(tc, ts);
    __syncthreads();

    long t = (long)blockIdx.x * 256 + threadIdx.x;   // NBC*64*16*8
    int kz = (int)(t & 7);
    int ky = (int)((t >> 3) & 15);
    long bcx = t >> 7;                               // bc*64 + x
    int kyact = (ky < 8) ? ky : (ky + 48);
    const float* zr = Zre + bcx * 512 + kz;
    const float* zi = Zim + bcx * 512 + kz;
    float sr = 0.f, si = 0.f;
    int id = 0;
#pragma unroll 16
    for (int y = 0; y < 64; ++y) {
        float c = tc[id], s = ts[id];
        float a = zr[y * 8], b = zi[y * 8];
        sr += a * c + b * s;      // (a+ib)(c - i s)
        si += b * c - a * s;
        id = (id + kyact) & 63;
    }
    Yre[t] = sr; Yim[t] = si;
}

// ---------------------------------------------------------------------------
// 4) Forward DFT along x (16 kept modes): X[bc][md], md = (kxi*16+kyi)*8+kz
// ---------------------------------------------------------------------------
__global__ __launch_bounds__(256) void fno_dftx_fwd(const float* __restrict__ Yre,
                                                    const float* __restrict__ Yim,
                                                    float* __restrict__ Xre,
                                                    float* __restrict__ Xim) {
    __shared__ float tc[64], ts[64];
    init_twiddle(tc, ts);
    __syncthreads();

    long t = (long)blockIdx.x * 256 + threadIdx.x;   // NBC*2048
    int md = (int)(t & 2047);
    long bc = t >> 11;
    int kxi = md >> 7, kyi = (md >> 3) & 15, kz = md & 7;
    int kxact = (kxi < 8) ? kxi : (kxi + 48);
    const float* yr = Yre + bc * 8192 + kyi * 8 + kz;
    const float* yi = Yim + bc * 8192 + kyi * 8 + kz;
    float sr = 0.f, si = 0.f;
    int id = 0;
#pragma unroll 16
    for (int x = 0; x < 64; ++x) {
        float c = tc[id], s = ts[id];
        float a = yr[x * 128], b = yi[x * 128];
        sr += a * c + b * s;
        si += b * c - a * s;
        id = (id + kxact) & 63;
    }
    Xre[t] = sr; Xim[t] = si;
}

// ---------------------------------------------------------------------------
// 5) Mode mixing (WMMA): per mode, complex GEMM [B=4 x Ci=32] x [Ci x Co=32].
//    scw layer base layout: [4 corners][Ci=32][Co=32][8][8][8][2].
//    One wave per mode; 8 x v_wmma_f32_16x16x32_f16 per mode.
// ---------------------------------------------------------------------------
__global__ __launch_bounds__(256) void fno_modemul(const float* __restrict__ Xre,
                                                   const float* __restrict__ Xim,
                                                   const float* __restrict__ scw,
                                                   float* __restrict__ Ore,
                                                   float* __restrict__ Oim) {
    int lane = threadIdx.x & 31, wave = threadIdx.x >> 5;
    int md = blockIdx.x * 8 + wave;                  // 0..2047
    int m = lane & 15, hi = lane >> 4;
    int kxi = md >> 7, kyi = (md >> 3) & 15, kz = md & 7;
    int corner = (kxi >= 8 ? 1 : 0) + (kyi >= 8 ? 2 : 0);
    int m512 = ((kxi & 7) * 8 + (kyi & 7)) * 8 + kz;
    const float* wbase = scw + (long)corner * 1048576L;   // 32*32*512*2

    // A tiles: rows = batch (4 valid of 16), K = ci.
    v16h ar, ai;
#pragma unroll
    for (int e = 0; e < 16; ++e) {
        int ci = kmapA(e, hi);
        if (m < 4) {
            long idx = ((long)m * 32 + ci) * 2048 + md;
            ar[e] = (_Float16)Xre[idx];
            ai[e] = (_Float16)Xim[idx];
        } else {
            ar[e] = (_Float16)0.0f;
            ai[e] = (_Float16)0.0f;
        }
    }

    v8f cre0 = {}, cre1 = {}, cim0 = {}, cim1 = {};
#pragma unroll
    for (int hf = 0; hf < 2; ++hf) {
        int co = hf * 16 + m;
        v16h br, bi, nbi;
#pragma unroll
        for (int e = 0; e < 16; ++e) {
            int ci = kmapB(e, hi);
            long widx = (((long)ci * 32 + co) * 512 + m512) * 2;
            float wr = wbase[widx];
            float wi = wbase[widx + 1];
            br[e]  = (_Float16)wr;
            bi[e]  = (_Float16)wi;
            nbi[e] = (_Float16)(-wi);
        }
        v8f& cr = hf ? cre1 : cre0;
        v8f& ci = hf ? cim1 : cim0;
        cr = wmma16(ar, br, cr);     // Re += Ar*Br
        cr = wmma16(ai, nbi, cr);    // Re -= Ai*Bi
        ci = wmma16(ar, bi, ci);     // Im += Ar*Bi
        ci = wmma16(ai, br, ci);     // Im += Ai*Br
    }

    // D: row r+8*hi = batch, col m = co within half.
#pragma unroll
    for (int r = 0; r < 8; ++r) {
        int b = r + 8 * hi;
        if (b < 4) {
            long o0 = ((long)b * 32 + m) * 2048 + md;
            long o1 = ((long)b * 32 + 16 + m) * 2048 + md;
            Ore[o0] = cre0[r]; Oim[o0] = cim0[r];
            Ore[o1] = cre1[r]; Oim[o1] = cim1[r];
        }
    }
}

// ---------------------------------------------------------------------------
// 6) Inverse DFT along x: G1[bc][x][ky][kz] = sum_kx O * e^{+i th kx x}
// ---------------------------------------------------------------------------
__global__ __launch_bounds__(256) void fno_dftx_inv(const float* __restrict__ Ore,
                                                    const float* __restrict__ Oim,
                                                    float* __restrict__ G1re,
                                                    float* __restrict__ G1im) {
    __shared__ float tc[64], ts[64];
    init_twiddle(tc, ts);
    __syncthreads();

    long t = (long)blockIdx.x * 256 + threadIdx.x;   // NBC*64*16*8
    int kz = (int)(t & 7);
    int ky = (int)((t >> 3) & 15);
    int x  = (int)((t >> 7) & 63);
    long bc = t >> 13;
    float sr = 0.f, si = 0.f;
#pragma unroll
    for (int kxi = 0; kxi < 16; ++kxi) {
        int kxact = (kxi < 8) ? kxi : (kxi + 48);
        int id = (kxact * x) & 63;
        float c = tc[id], s = ts[id];
        long idx = bc * 2048 + (long)(kxi * 16 + ky) * 8 + kz;
        float a = Ore[idx], b = Oim[idx];
        sr += a * c - b * s;      // (a+ib)(c + i s)
        si += a * s + b * c;
    }
    G1re[t] = sr; G1im[t] = si;
}

// ---------------------------------------------------------------------------
// 7) Inverse DFT along y: G2[bc][x][y][kz] = sum_ky G1 * e^{+i th ky y}
// ---------------------------------------------------------------------------
__global__ __launch_bounds__(256) void fno_dfty_inv(const float* __restrict__ G1re,
                                                    const float* __restrict__ G1im,
                                                    float* __restrict__ G2re,
                                                    float* __restrict__ G2im) {
    __shared__ float tc[64], ts[64];
    init_twiddle(tc, ts);
    __syncthreads();

    long t = (long)blockIdx.x * 256 + threadIdx.x;   // NBC*64*64*8
    int kz = (int)(t & 7);
    int y  = (int)((t >> 3) & 63);
    int x  = (int)((t >> 9) & 63);
    long bc = t >> 15;
    float sr = 0.f, si = 0.f;
#pragma unroll
    for (int kyi = 0; kyi < 16; ++kyi) {
        int kyact = (kyi < 8) ? kyi : (kyi + 48);
        int id = (kyact * y) & 63;
        float c = tc[id], s = ts[id];
        long idx = ((bc * 64 + x) * 16 + kyi) * 8 + kz;
        float a = G1re[idx], b = G1im[idx];
        sr += a * c - b * s;
        si += a * s + b * c;
    }
    G2re[t] = sr; G2im[t] = si;
}

// ---------------------------------------------------------------------------
// 8) Pointwise conv (WMMA) fused with inverse real z-DFT + bias + (GELU):
//    out[b][co][p] = act( irfft_z(G2)[b][co][p] + sum_ci w[co][ci]*h[b][ci][p]
//                         + conv_b[co] )
//    A = weights (M=co 16, K=ci 32) from LDS; B = activations (K=ci, N=16 pts).
//    A 16-point tile is 16 consecutive z in one (x,y) row, so the 8 complex
//    G2 coefficients per co are lane-broadcast loads; z-twiddles from LDS.
// ---------------------------------------------------------------------------
__global__ __launch_bounds__(256) void fno_pointwise(const float* __restrict__ h,
                                                     const float* __restrict__ G2re,
                                                     const float* __restrict__ G2im,
                                                     const float* __restrict__ w,
                                                     const float* __restrict__ bias,
                                                     float* __restrict__ out,
                                                     int apply_gelu) {
    __shared__ float lw[1024];
    __shared__ float lb[32];
    __shared__ float tc[64], ts[64];
    int t = threadIdx.x;
#pragma unroll
    for (int i = 0; i < 4; ++i) lw[t + 256 * i] = w[t + 256 * i];
    if (t < 32) lb[t] = bias[t];
    init_twiddle(tc, ts);
    __syncthreads();

    int lane = t & 31, wave = t >> 5;
    int m = lane & 15, hi = lane >> 4;

    // A tiles (weights) from LDS: A[m][k] = w[(co0+m)*32 + k]
    v16h a0, a1;
#pragma unroll
    for (int e = 0; e < 16; ++e) {
        int k = kmapA(e, hi);
        a0[e] = (_Float16)lw[m * 32 + k];
        a1[e] = (_Float16)lw[(16 + m) * 32 + k];
    }

    long b = blockIdx.x >> 9;                         // 512 blocks / batch
    long pblk = (long)(blockIdx.x & 511) * 512 + (long)wave * 64;
    const float* hb  = h + b * 32 * PVOL;
    float* ob        = out + b * 32 * PVOL;
    const float scale = 1.0f / 262144.0f;             // 1/64^3

#pragma unroll
    for (int tile = 0; tile < 4; ++tile) {
        long p0 = pblk + (long)tile * 16;
        long rowxy = p0 >> 6;                         // (x*64+y), uniform
        int z = (int)(p0 & 63) + m;                   // per-lane z (same row)

        // Per-lane z-twiddles for kz = 1..7.
        float ck[8], sk[8];
#pragma unroll
        for (int kz = 1; kz < 8; ++kz) {
            int id = (kz * z) & 63;
            ck[kz] = tc[id];
            sk[kz] = ts[id];
        }

        // B tile: K=ci, N=point (lane m -> point p0+m).
        v16h bm;
#pragma unroll
        for (int e = 0; e < 16; ++e) {
            int k = kmapB(e, hi);
            bm[e] = (_Float16)hb[(long)k * PVOL + p0 + m];
        }
        v8f c0 = {}, c1 = {};
        c0 = wmma16(a0, bm, c0);
        c1 = wmma16(a1, bm, c1);

#pragma unroll
        for (int r = 0; r < 8; ++r) {
            int co = r + 8 * hi;                      // within first 16-co tile
            long pp = p0 + m;

            // irfft-z from the row's 8 complex coefficients (lane-broadcast).
            const float* gr0 = G2re + (((long)b * 32 + co) * 4096 + rowxy) * 8;
            const float* gi0 = G2im + (((long)b * 32 + co) * 4096 + rowxy) * 8;
            const float* gr1 = gr0 + 16L * 4096 * 8;  // co+16
            const float* gi1 = gi0 + 16L * 4096 * 8;
            float xr0 = gr0[0];
            float xr1 = gr1[0];
#pragma unroll
            for (int kz = 1; kz < 8; ++kz) {
                xr0 += 2.0f * (gr0[kz] * ck[kz] - gi0[kz] * sk[kz]);
                xr1 += 2.0f * (gr1[kz] * ck[kz] - gi1[kz] * sk[kz]);
            }

            float v0 = c0[r] + lb[co]      + xr0 * scale;
            float v1 = c1[r] + lb[co + 16] + xr1 * scale;
            if (apply_gelu) {
                v0 = 0.5f * v0 * (1.0f + erff(v0 * 0.70710678f));
                v1 = 0.5f * v1 * (1.0f + erff(v1 * 0.70710678f));
            }
            ob[(long)co * PVOL + pp]        = v0;
            ob[(long)(co + 16) * PVOL + pp] = v1;
        }
    }
}

// ---------------------------------------------------------------------------
// 9) Projection: out[b][p] = q_b + sum_co h[b][co][p] * q_w[co]
// ---------------------------------------------------------------------------
__global__ __launch_bounds__(256) void fno_project(const float* __restrict__ h,
                                                   const float* __restrict__ q_w,
                                                   const float* __restrict__ q_b,
                                                   float* __restrict__ out) {
    long t = (long)blockIdx.x * 256 + threadIdx.x;   // B*P
    long b = t >> 18;
    long p = t & (PVOL - 1);
    const float* hb = h + b * 32 * PVOL + p;
    float acc = q_b[0];
#pragma unroll
    for (int co = 0; co < 32; ++co) acc += hb[(long)co * PVOL] * q_w[co];
    out[t] = acc;
}

// ---------------------------------------------------------------------------
// Host-side launcher
// ---------------------------------------------------------------------------
extern "C" void kernel_launch(void* const* d_in, const int* in_sizes, int n_in,
                              void* d_out, int out_size, void* d_ws, size_t ws_size,
                              hipStream_t stream) {
    const float* xin    = (const float*)d_in[0];   // [B,64,64,64,1]
    const float* p_w    = (const float*)d_in[1];   // [4,32]
    const float* p_b    = (const float*)d_in[2];   // [32]
    const float* scw    = (const float*)d_in[3];   // [L,4,32,32,8,8,8,2]
    const float* conv_w = (const float*)d_in[4];   // [L,32,32]
    const float* conv_b = (const float*)d_in[5];   // [L,32]
    const float* q_w    = (const float*)d_in[6];   // [32,1]
    const float* q_b    = (const float*)d_in[7];   // [1]
    float* out = (float*)d_out;

    // Workspace carving (floats).
    float* ws = (float*)d_ws;
    float* hA   = ws;                 // 33,554,432
    float* hB   = hA + 33554432L;     // 33,554,432
    float* SAre = hB + 33554432L;     // 4,194,304  (Z fwd / G2 inv)
    float* SAim = SAre + 4194304L;
    float* SBre = SAim + 4194304L;    // 1,048,576  (Y fwd / G1 inv)
    float* SBim = SBre + 1048576L;
    float* Xre  = SBim + 1048576L;    // 262,144
    float* Xim  = Xre + 262144L;
    float* Ore  = Xim + 262144L;      // 262,144
    float* Oim  = Ore + 262144L;

    dim3 blk(256);

    // Lift: x + coordinate grid -> 32 channels.
    fno_lift<<<dim3(4096), blk, 0, stream>>>(xin, p_w, p_b, hA);

    float* cur = hA;
    float* nxt = hB;
    for (int l = 0; l < 4; ++l) {
        const float* scw_l = scw + (long)l * 4194304L;   // 4*32*32*512*2
        const float* cw_l  = conv_w + (long)l * 1024L;
        const float* cb_l  = conv_b + (long)l * 32L;

        fno_dftz_fwd<<<dim3(16384), blk, 0, stream>>>(cur, SAre, SAim);
        fno_dfty_fwd<<<dim3(4096),  blk, 0, stream>>>(SAre, SAim, SBre, SBim);
        fno_dftx_fwd<<<dim3(1024),  blk, 0, stream>>>(SBre, SBim, Xre, Xim);
        fno_modemul <<<dim3(256),   blk, 0, stream>>>(Xre, Xim, scw_l, Ore, Oim);
        fno_dftx_inv<<<dim3(4096),  blk, 0, stream>>>(Ore, Oim, SBre, SBim);
        fno_dfty_inv<<<dim3(16384), blk, 0, stream>>>(SBre, SBim, SAre, SAim);
        fno_pointwise<<<dim3(2048), blk, 0, stream>>>(cur, SAre, SAim,
                                                      cw_l, cb_l, nxt,
                                                      (l < 3) ? 1 : 0);
        float* tmp = cur; cur = nxt; nxt = tmp;
    }

    fno_project<<<dim3(4096), blk, 0, stream>>>(cur, q_w, q_b, out);
}
